// POLY2_90031104458808
// MI455X (gfx1250) — matrix operand order repaired and checked by reading, not compiled, not run on hardware
//
#include <hip/hip_runtime.h>

#define F 2048
#define BATCH 16384
#define KC 4  // k-steps (32 each) kept in registers per chunk

typedef __attribute__((ext_vector_type(16))) __bf16 v16bf;
typedef __attribute__((ext_vector_type(8)))  float  v8f;

union BF16Frag {
    v16bf v;
    unsigned short u[16];
    uint4 q[2];
};

__device__ __forceinline__ unsigned short f2bf_rne(float f) {
    unsigned int u = __float_as_uint(f);
    unsigned int r = u + 0x7FFFu + ((u >> 16) & 1u);
    return (unsigned short)(r >> 16);
}
__device__ __forceinline__ float bf2f(unsigned short h) {
    return __uint_as_float(((unsigned int)h) << 16);
}

// ---------------------------------------------------------------------------
// Kernel 1: scatter bw (strict upper triangle, row-major triu k=1 order) into
// a dense TRANSPOSED matrix Bt[n][k] = B[k][n], split into bf16 hi/lo planes.
// Transposed storage makes the WMMA B-fragment (per-lane K run) contiguous.
// ---------------------------------------------------------------------------
__global__ __launch_bounds__(256) void poly2_build_bt(
    const float* __restrict__ bw,
    unsigned short* __restrict__ bt_hi,
    unsigned short* __restrict__ bt_lo) {
    int id = blockIdx.x * 256 + threadIdx.x;
    if (id >= F * F) return;
    int n = id >> 11;        // column of B == row of Bt
    int k = id & (F - 1);    // row of B    == col of Bt (contiguous per thread)
    float val = 0.0f;
    if (k < n) {
        // pairs before row k: k*(F-1) - k*(k-1)/2 ; within row: (n - k - 1)
        int idx = k * (F - 1) - ((k * (k - 1)) >> 1) + (n - k - 1);
        val = bw[idx];
    }
    unsigned short h = f2bf_rne(val);
    unsigned short l = f2bf_rne(val - bf2f(h));
    bt_hi[id] = h;   // id == n*F + k exactly
    bt_lo[id] = l;
}

// ---------------------------------------------------------------------------
// Kernel 2: fused  cross[b] = x[b] · (X@B)[b],  lin[b] = x[b]·w,
//           out[b] = sigmoid(w0 + lin + cross)
// One wave per 16 batch rows; k-outer / n-inner so A-frags (bf16 hi/lo split,
// ~f32 accuracy via 3 WMMAs) live in registers across the whole N sweep.
// ---------------------------------------------------------------------------
__global__ __launch_bounds__(256) void poly2_main(
    const float* __restrict__ x,
    const float* __restrict__ w0p,
    const float* __restrict__ w,
    const unsigned short* __restrict__ bt_hi,
    const unsigned short* __restrict__ bt_lo,
    float* __restrict__ out) {
    const int lane = threadIdx.x & 31;
    const int wave = threadIdx.x >> 5;
    const int m0   = (blockIdx.x * 8 + wave) * 16;   // 16 batch rows per wave
    const int r    = lane & 15;                      // row (A) / col (B,D) index
    const int hf   = lane >> 4;                      // lane half

    const float* xrow = x + (size_t)(m0 + r) * F;

    float cross[8];
#pragma unroll
    for (int v = 0; v < 8; ++v) cross[v] = 0.0f;
    float lin = 0.0f;

#pragma unroll 1
    for (int kc = 0; kc < F; kc += 32 * KC) {
        // ---- build A fragments (contiguous k = 16*hf + s, matching B) + fused x.w ----
        BF16Frag Ahi[KC], Alo[KC];
#pragma unroll
        for (int kk = 0; kk < KC; ++kk) {
            const int k0 = kc + kk * 32;
            float a[16], wv[16];
            const float4* p0 = reinterpret_cast<const float4*>(xrow + k0 + hf * 16);
            const float4* p1 = reinterpret_cast<const float4*>(xrow + k0 + hf * 16 + 8);
            *reinterpret_cast<float4*>(&a[0])  = p0[0];
            *reinterpret_cast<float4*>(&a[4])  = p0[1];
            *reinterpret_cast<float4*>(&a[8])  = p1[0];
            *reinterpret_cast<float4*>(&a[12]) = p1[1];
            const float4* q0 = reinterpret_cast<const float4*>(w + k0 + hf * 16);
            const float4* q1 = reinterpret_cast<const float4*>(w + k0 + hf * 16 + 8);
            *reinterpret_cast<float4*>(&wv[0])  = q0[0];
            *reinterpret_cast<float4*>(&wv[4])  = q0[1];
            *reinterpret_cast<float4*>(&wv[8])  = q1[0];
            *reinterpret_cast<float4*>(&wv[12]) = q1[1];
#pragma unroll
            for (int i = 0; i < 16; ++i) {
                lin = __builtin_fmaf(a[i], wv[i], lin);
                unsigned short h = f2bf_rne(a[i]);
                Ahi[kk].u[i] = h;
                Alo[kk].u[i] = f2bf_rne(a[i] - bf2f(h));
            }
        }

        // ---- sweep all N columns with A resident in registers ----
#pragma unroll 2
        for (int n0 = 0; n0 < F; n0 += 16) {
            const int col = n0 + r;
            // prefetch next column's B stream (speculative; OOB is dropped)
            __builtin_prefetch(bt_hi + (size_t)(col + 16) * F + kc, 0, 1);

            // Hoist the row-dot multipliers: issue all 8 loads up front so
            // their latency hides under the B-loads + WMMA chain below.
            // D layout: lane l, VGPR v -> row = v + 8*(l>>4), col = l&15
            float xm[8];
#pragma unroll
            for (int v = 0; v < 8; ++v)
                xm[v] = x[(size_t)(m0 + v + 8 * hf) * F + col];

            // Three independent accumulator chains (hi*hi, hi*lo, lo*hi)
            // to keep the XDL pipe busy instead of one serial D->C chain.
            v8f acc0 = {0.f, 0.f, 0.f, 0.f, 0.f, 0.f, 0.f, 0.f};
            v8f acc1 = {0.f, 0.f, 0.f, 0.f, 0.f, 0.f, 0.f, 0.f};
            v8f acc2 = {0.f, 0.f, 0.f, 0.f, 0.f, 0.f, 0.f, 0.f};
#pragma unroll
            for (int kk = 0; kk < KC; ++kk) {
                const size_t boff = (size_t)col * F + kc + kk * 32 + hf * 16;
                BF16Frag Bhi, Blo;
                const uint4* bh = reinterpret_cast<const uint4*>(bt_hi + boff);
                Bhi.q[0] = bh[0];
                Bhi.q[1] = bh[1];
                const uint4* bl = reinterpret_cast<const uint4*>(bt_lo + boff);
                Blo.q[0] = bl[0];
                Blo.q[1] = bl[1];
                acc0 = __builtin_amdgcn_wmma_f32_16x16x32_bf16(
                    false, Ahi[kk].v, false, Bhi.v, (short)0, acc0, false, false);
                acc1 = __builtin_amdgcn_wmma_f32_16x16x32_bf16(
                    false, Ahi[kk].v, false, Blo.v, (short)0, acc1, false, false);
                acc2 = __builtin_amdgcn_wmma_f32_16x16x32_bf16(
                    false, Alo[kk].v, false, Bhi.v, (short)0, acc2, false, false);
            }
            v8f ys = (acc0 + acc1) + acc2;
            // fuse row-wise dot:  cross[row] += Y[row,col] * X[row,col]
#pragma unroll
            for (int v = 0; v < 8; ++v)
                cross[v] = __builtin_fmaf(ys[v], xm[v], cross[v]);
        }
    }

    // ---- wave32 reductions ----
    lin += __shfl_xor(lin, 16, 32);   // combine the two K-halves of each row
#pragma unroll
    for (int v = 0; v < 8; ++v) {     // sum the 16 column classes per half
        cross[v] += __shfl_xor(cross[v], 8, 32);
        cross[v] += __shfl_xor(cross[v], 4, 32);
        cross[v] += __shfl_xor(cross[v], 2, 32);
        cross[v] += __shfl_xor(cross[v], 1, 32);
    }

    const float w0v = w0p[0];
#pragma unroll
    for (int v = 0; v < 8; ++v) {
        // linear term for row m0+j lives on lane j (both halves after xor-16)
        float lv = __shfl(lin, v + 8 * hf, 32);
        float z  = w0v + lv + cross[v];
        float s  = 1.0f / (1.0f + __expf(-z));
        if (r == 0) out[m0 + v + 8 * hf] = s;
    }
}

// ---------------------------------------------------------------------------
extern "C" void kernel_launch(void* const* d_in, const int* in_sizes, int n_in,
                              void* d_out, int out_size, void* d_ws, size_t ws_size,
                              hipStream_t stream) {
    const float* x  = (const float*)d_in[0];   // [16384, 2048]
    const float* w0 = (const float*)d_in[1];   // [1, 1]
    const float* w  = (const float*)d_in[2];   // [2048, 1]
    const float* bw = (const float*)d_in[3];   // [N_PAIRS, 1]
    float* out = (float*)d_out;                // [16384, 1]

    unsigned short* bt_hi = (unsigned short*)d_ws;          // 8 MiB
    unsigned short* bt_lo = bt_hi + (size_t)F * F;          // 8 MiB

    poly2_build_bt<<<(F * F) / 256, 256, 0, stream>>>(bw, bt_hi, bt_lo);
    poly2_main<<<BATCH / 128, 256, 0, stream>>>(x, w0, w, bt_hi, bt_lo, out);
}
